// AmbientReflectionNet_74294344286346
// MI455X (gfx1250) — matrix operand order repaired and checked by reading, not compiled, not run on hardware
//
#include <hip/hip_runtime.h>

typedef __attribute__((ext_vector_type(16))) _Float16 v16h;
typedef __attribute__((ext_vector_type(8)))  float    v8f;

#define P_PTS   262144
#define PTS_PER_BLOCK 32              // 2 M-tiles of 16 points
#define NPB     (P_PTS / PTS_PER_BLOCK)   // 8192 point-blocks per MLP
#define EPSN    1e-12f

// Packed weight layout in d_ws (halfs). Tile = 512 halfs (32 lanes x 16 halfs = 1KB).
// Matrix order: dW0,dW1,dW2,dW3,sW0,sW1,sW2,sW3
// Tile counts :  16, 128, 128,  8,  16, 128, 128,  8   (Kc x Nt)
#define WOFF_D0 0u
#define WOFF_D1 8192u
#define WOFF_D2 73728u
#define WOFF_D3 139264u
#define WOFF_S0 143360u
#define WOFF_S1 151552u
#define WOFF_S2 217088u
#define WOFF_S3 282624u
#define WPK_HALFS (560u * 512u)       // 286720 halfs = 560 KB

// ---------------------------------------------------------------------------
// Kernel 1: convert weights to f16 pre-swizzled into the WMMA B-operand
// register layout (dense B 32x16: lanes 0-15 hold K=0..7,16..23, lanes 16-31 hold
// K=8..15,24..31, column = lane%16, half index h <-> K within lane group).
// Zero-pads K (3->32, 8->32) and N (3->16) so layers 0 and 3 are WMMAs too.
// ---------------------------------------------------------------------------
__global__ void __launch_bounds__(256) pack_weights_kernel(
    const float* dW0, const float* dW1, const float* dW2, const float* dW3,
    const float* sW0, const float* sW1, const float* sW2, const float* sW3,
    _Float16* wpk)
{
    const unsigned tid = blockIdx.x * blockDim.x + threadIdx.x;
    if (tid >= WPK_HALFS) return;

    const int tile = tid >> 9;
    const int r    = tid & 511;
    const int lane = r >> 4;
    const int h    = r & 15;

    const int   cum[9]  = {0, 16, 144, 272, 280, 296, 424, 552, 560};
    const int   Nt[8]   = {16, 16, 16, 1, 16, 16, 16, 1};
    const int   Kact[8] = {3, 256, 256, 256, 8, 256, 256, 256};
    const int   Nact[8] = {256, 256, 256, 3, 256, 256, 256, 3};
    const float* src[8] = {dW0, dW1, dW2, dW3, sW0, sW1, sW2, sW3};

    int m = 0;
    while (tile >= cum[m + 1]) ++m;
    const int lt = tile - cum[m];
    const int kc = lt / Nt[m];
    const int nt = lt % Nt[m];

    const int n = nt * 16 + (lane & 15);
    const int k = kc * 32 + (h < 8 ? h : h + 8) + 8 * (lane >> 4);    // dense B layout: K = (h<8?h:h+8) + 8*hi

    float v = 0.0f;
    if (k < Kact[m] && n < Nact[m]) v = src[m][k * Nact[m] + n];
    wpk[tid] = (_Float16)v;
}

// ---------------------------------------------------------------------------
// Kernel 2: fused 4-layer MLP. One block (8 waves) per 32 points (2 M-tiles)
// per MLP. Each wave owns a 32-wide N slice; each loaded B tile is reused for
// both M-tiles (2x weight-traffic reduction vs 16-pt blocks). Activations in
// LDS in the WMMA A-operand layout (K = (h<8?h:h+8)+8*hi, row = lane%16),
// double-buffered.
// ---------------------------------------------------------------------------
__global__ void __launch_bounds__(256) mlp_fused_kernel(
    const float* __restrict__ normals, const float* __restrict__ view_dirs,
    const float* __restrict__ rough,   const float* __restrict__ r0,
    const float* db0, const float* db1, const float* db2, const float* db3,
    const float* sb0, const float* sb1, const float* sb2, const float* sb3,
    const _Float16* __restrict__ wpk,
    float* __restrict__ out)
{
    __shared__ __align__(32) _Float16 act[2][2][4096]; // [buf][mtile][8 chunks x 512]
    __shared__ float smask[PTS_PER_BLOCK];

    const int bid  = blockIdx.x;
    const int spec = (bid >= NPB) ? 1 : 0;
    const int tile = spec ? (bid - NPB) : bid;
    const int wave = threadIdx.x >> 5;
    const int lane = threadIdx.x & 31;
    const int nlo  = lane & 15;

    // ---- build layer-0 A tiles (K padded to 32) + visibility mask ----------
    if (wave < 2) {
        const int mt = wave;
        const int m  = lane & 15;
        const int p  = tile * PTS_PER_BLOCK + mt * 16 + m;
        _Float16 h8[8] = {};
        if (lane < 16) {
            float nx = normals[p * 3 + 0], ny = normals[p * 3 + 1], nz = normals[p * 3 + 2];
            float vx = view_dirs[p * 3 + 0], vy = view_dirs[p * 3 + 1], vz = view_dirs[p * 3 + 2];
            float nn = fmaxf(sqrtf(nx * nx + ny * ny + nz * nz), EPSN);
            float vn = fmaxf(sqrtf(vx * vx + vy * vy + vz * vz), EPSN);
            nx /= nn; ny /= nn; nz /= nn;
            vx /= vn; vy /= vn; vz /= vn;
            const float dot = nx * vx + ny * vy + nz * vz;
            smask[mt * 16 + m] = (dot > 0.0f) ? 1.0f : 0.0f;
            h8[0] = (_Float16)nx; h8[1] = (_Float16)ny; h8[2] = (_Float16)nz;
            if (spec) {
                h8[3] = (_Float16)vx; h8[4] = (_Float16)vy; h8[5] = (_Float16)vz;
                h8[6] = (_Float16)rough[p]; h8[7] = (_Float16)r0[p];
            }
        }
#pragma unroll
        for (int h = 0; h < 16; ++h) {
            _Float16 val = (lane < 16 && h < 8) ? h8[h] : (_Float16)0.0f;
            act[0][mt][lane * 16 + h] = val;
        }
    }
    __syncthreads();

    const unsigned wofsD[4] = {WOFF_D0, WOFF_D1, WOFF_D2, WOFF_D3};
    const unsigned wofsS[4] = {WOFF_S0, WOFF_S1, WOFF_S2, WOFF_S3};
    const float* biasD[4] = {db0, db1, db2, db3};
    const float* biasS[4] = {sb0, sb1, sb2, sb3};

    // C-tile -> A-tile scatter base (offsets v*16 + t*8 within chunk)
    const int halfbase = ((lane >> 3) & 1) * 256 + (lane >> 4) * 128 + (lane & 7);

    int cur = 0;
    // ---- hidden layers: each wave owns N slice [wave*32, wave*32+32) -------
#pragma unroll
    for (int l = 0; l < 3; ++l) {
        const _Float16* W = wpk + (spec ? wofsS[l] : wofsD[l]);
        const float* bias = spec ? biasS[l] : biasD[l];
        const int KC      = (l == 0) ? 1 : 8;
        const int nt0     = wave * 2;

        v8f acc[2][2] = {};                         // [mtile][ntile]
        for (int kc = 0; kc < KC; ++kc) {
            v16h a0 = *(const v16h*)&act[cur][0][kc * 512 + lane * 16];
            v16h a1 = *(const v16h*)&act[cur][1][kc * 512 + lane * 16];
            v16h b0 = *(const v16h*)&W[(kc * 16 + nt0) * 512 + lane * 16];
            v16h b1 = *(const v16h*)&W[(kc * 16 + nt0 + 1) * 512 + lane * 16];
            if (kc + 1 < KC)
                __builtin_prefetch(&W[((kc + 1) * 16 + nt0) * 512 + lane * 16], 0, 3);
            acc[0][0] = __builtin_amdgcn_wmma_f32_16x16x32_f16(false, a0, false, b0,
                                                               (short)0, acc[0][0], false, false);
            acc[0][1] = __builtin_amdgcn_wmma_f32_16x16x32_f16(false, a0, false, b1,
                                                               (short)0, acc[0][1], false, false);
            acc[1][0] = __builtin_amdgcn_wmma_f32_16x16x32_f16(false, a1, false, b0,
                                                               (short)0, acc[1][0], false, false);
            acc[1][1] = __builtin_amdgcn_wmma_f32_16x16x32_f16(false, a1, false, b1,
                                                               (short)0, acc[1][1], false, false);
        }

        const float bv0 = bias[nt0 * 16 + nlo];
        const float bv1 = bias[nt0 * 16 + 16 + nlo];
        const int nxt = cur ^ 1;
#pragma unroll
        for (int mt = 0; mt < 2; ++mt) {
            _Float16* dst = &act[nxt][mt][wave * 512 + halfbase];
#pragma unroll
            for (int v = 0; v < 8; ++v) {
                float x0 = fmaxf(acc[mt][0][v] + bv0, 0.0f);
                float x1 = fmaxf(acc[mt][1][v] + bv1, 0.0f);
                dst[v * 16]     = (_Float16)x0;   // t=0 half
                dst[v * 16 + 8] = (_Float16)x1;   // t=1 half
            }
        }
        cur = nxt;
        __syncthreads();
    }

    // ---- final layer: 256 -> 16 (cols 0..2 valid), one M-tile per wave -----
    if (wave < 2) {
        const int mt = wave;
        const _Float16* W3 = wpk + (spec ? WOFF_S3 : WOFF_D3);
        const float* b3    = spec ? sb3 : db3;
        v8f acc = {};
#pragma unroll
        for (int kc = 0; kc < 8; ++kc) {
            v16h a = *(const v16h*)&act[cur][mt][kc * 512 + lane * 16];
            v16h b = *(const v16h*)&W3[kc * 512 + lane * 16];
            acc = __builtin_amdgcn_wmma_f32_16x16x32_f16(false, a, false, b,
                                                         (short)0, acc, false, false);
        }
        const int n = nlo;
        if (n < 3) {
            const float bv = b3[n];
            float* obase = out + (spec ? (P_PTS * 3) : 0);
#pragma unroll
            for (int v = 0; v < 8; ++v) {
                const int m = v + 8 * (lane >> 4);
                const float val = (acc[v] + bv) * smask[mt * 16 + m];
                obase[(tile * PTS_PER_BLOCK + mt * 16 + m) * 3 + n] = val;
            }
        }
    }
}

// ---------------------------------------------------------------------------
extern "C" void kernel_launch(void* const* d_in, const int* in_sizes, int n_in,
                              void* d_out, int out_size, void* d_ws, size_t ws_size,
                              hipStream_t stream) {
    const float* normals   = (const float*)d_in[0];
    const float* view_dirs = (const float*)d_in[1];
    const float* rough     = (const float*)d_in[2];
    const float* r0        = (const float*)d_in[3];
    const float* dW0 = (const float*)d_in[4];  const float* db0 = (const float*)d_in[5];
    const float* dW1 = (const float*)d_in[6];  const float* db1 = (const float*)d_in[7];
    const float* dW2 = (const float*)d_in[8];  const float* db2 = (const float*)d_in[9];
    const float* dW3 = (const float*)d_in[10]; const float* db3 = (const float*)d_in[11];
    const float* sW0 = (const float*)d_in[12]; const float* sb0 = (const float*)d_in[13];
    const float* sW1 = (const float*)d_in[14]; const float* sb1 = (const float*)d_in[15];
    const float* sW2 = (const float*)d_in[16]; const float* sb2 = (const float*)d_in[17];
    const float* sW3 = (const float*)d_in[18]; const float* sb3 = (const float*)d_in[19];

    _Float16* wpk = (_Float16*)d_ws;   // 560 KB packed f16 weights
    float* out    = (float*)d_out;

    pack_weights_kernel<<<(WPK_HALFS + 255) / 256, 256, 0, stream>>>(
        dW0, dW1, dW2, dW3, sW0, sW1, sW2, sW3, wpk);

    mlp_fused_kernel<<<2 * NPB, 256, 0, stream>>>(
        normals, view_dirs, rough, r0,
        db0, db1, db2, db3, sb0, sb1, sb2, sb3,
        wpk, out);
}
